// LinearLowbit_77043123355779
// MI455X (gfx1250) — 8-bit matrix operand order repaired and checked by reading, compiled, not run on hardware
//
#include <hip/hip_runtime.h>
#include <cstdint>
#include <cstddef>

typedef int   v16i __attribute__((ext_vector_type(16)));
typedef float v8f  __attribute__((ext_vector_type(8)));

#define M_TOT 8192
#define N_TOT 4096
#define K_TOT 4096

__global__ void lb_absum_partial(const float* __restrict__ x,
                                 float* __restrict__ part, int n4) {
    int tid = threadIdx.x;
    int i = blockIdx.x * blockDim.x + tid;
    int stride = gridDim.x * blockDim.x;
    const float4* x4 = (const float4*)x;
    float acc = 0.0f;
    for (; i < n4; i += stride) {
        float4 v = x4[i];
        acc += fabsf(v.x) + fabsf(v.y) + fabsf(v.z) + fabsf(v.w);
    }
    __shared__ float sm[256];
    sm[tid] = acc;
    __syncthreads();
    for (int off = 128; off > 0; off >>= 1) {
        if (tid < off) sm[tid] += sm[tid + off];
        __syncthreads();
    }
    if (tid == 0) part[blockIdx.x] = sm[0];
}

__global__ void lb_finalize_scale(const float* __restrict__ part,
                                  float* __restrict__ dst, float invN) {
    int tid = threadIdx.x;
    float acc = part[tid] + part[tid + 256] + part[tid + 512] + part[tid + 768];
    __shared__ float sm[256];
    sm[tid] = acc;
    __syncthreads();
    for (int off = 128; off > 0; off >>= 1) {
        if (tid < off) sm[tid] += sm[tid + off];
        __syncthreads();
    }
    if (tid == 0) dst[0] = sm[0] * invN;
}

__device__ __forceinline__ unsigned int lb_q1(float x, float inv_s) {
    float y = x * inv_s;
    float a = fabsf(y);
    int idx = (a >= 0.25f) + (a >= 0.75f) + (a >= 1.25f) + (a >= 1.75f) +
              (a >= 2.5f)  + (a >= 3.5f)  + (a >= 5.0f);
    const unsigned long long lut = 0x4C4844403C383000ULL;
    unsigned int b = (unsigned int)((lut >> (idx * 8)) & 0xFFu);
    b |= (__float_as_uint(y) >> 31) << 7;
    return b;
}

__global__ void lb_quant_fp8(const float* __restrict__ x,
                             unsigned char* __restrict__ dst,
                             const float* __restrict__ scalePtr, int n4) {
    int i = blockIdx.x * blockDim.x + threadIdx.x;
    if (i >= n4) return;
    float inv = 1.0f / scalePtr[0];
    float4 v = ((const float4*)x)[i];
    unsigned int b0 = lb_q1(v.x, inv);
    unsigned int b1 = lb_q1(v.y, inv);
    unsigned int b2 = lb_q1(v.z, inv);
    unsigned int b3 = lb_q1(v.w, inv);
    ((unsigned int*)dst)[i] = b0 | (b1 << 8) | (b2 << 16) | (b3 << 24);
}

__device__ __forceinline__ v16i lb_loadB(const unsigned char* row, int k0, int h) {
    v16i b;
#pragma unroll
    for (int q = 0; q < 4; ++q) {
        const int4 t = *(const int4*)(row + k0 + 32 * q + 16 * h);
        b[4 * q + 0] = t.x; b[4 * q + 1] = t.y;
        b[4 * q + 2] = t.z; b[4 * q + 3] = t.w;
    }
    return b;
}

__global__ void __launch_bounds__(256)
lb_gemm_fp8(const unsigned char* __restrict__ A8,
            const unsigned char* __restrict__ W8,
            const float* __restrict__ bias,
            const float* __restrict__ scales,
            float* __restrict__ out) {
    const int tid  = threadIdx.x;
    const int wave = tid >> 5;
    const int lane = tid & 31;
    const int l15  = lane & 15;
    const int h    = lane >> 4;

    const int wid    = blockIdx.x * 8 + wave;
    const int nGroup = wid & 63;
    const int mTile  = wid >> 6;

    const unsigned char* aRow = A8 + (size_t)(mTile * 16 + l15) * K_TOT;
    const unsigned char* bRow0 = W8 + (size_t)(nGroup * 64 +  0 + l15) * K_TOT;
    const unsigned char* bRow1 = W8 + (size_t)(nGroup * 64 + 16 + l15) * K_TOT;
    const unsigned char* bRow2 = W8 + (size_t)(nGroup * 64 + 32 + l15) * K_TOT;
    const unsigned char* bRow3 = W8 + (size_t)(nGroup * 64 + 48 + l15) * K_TOT;

    v8f acc0 = {0.f,0.f,0.f,0.f,0.f,0.f,0.f,0.f};
    v8f acc1 = acc0, acc2 = acc0, acc3 = acc0;

    for (int kk = 0; kk < K_TOT / 128; ++kk) {
        const int k0 = kk << 7;
        if (kk + 1 < K_TOT / 128) {
            __builtin_prefetch(aRow + k0 + 128, 0, 1);
        }
        v16i a;
#pragma unroll
        for (int j = 0; j < 8; ++j) {
            const uint2 t = *(const uint2*)(aRow + k0 + 32 * (j >> 1) + 16 * h + 8 * (j & 1));
            const int di = ((j >> 2) << 3) + ((j & 3) << 1);
            a[di + 0] = (int)t.x;
            a[di + 1] = (int)t.y;
        }
        acc0 = __builtin_amdgcn_wmma_f32_16x16x128_fp8_fp8(
                   a, lb_loadB(bRow0, k0, h), (short)0, acc0, false, false);
        acc1 = __builtin_amdgcn_wmma_f32_16x16x128_fp8_fp8(
                   a, lb_loadB(bRow1, k0, h), (short)0, acc1, false, false);
        acc2 = __builtin_amdgcn_wmma_f32_16x16x128_fp8_fp8(
                   a, lb_loadB(bRow2, k0, h), (short)0, acc2, false, false);
        acc3 = __builtin_amdgcn_wmma_f32_16x16x128_fp8_fp8(
                   a, lb_loadB(bRow3, k0, h), (short)0, acc3, false, false);
    }

    const float s = scales[0] * scales[1];
    const int mBase = mTile * 16 + h * 8;

    v8f accs[4] = {acc0, acc1, acc2, acc3};
#pragma unroll
    for (int t = 0; t < 4; ++t) {
        const int n = nGroup * 64 + t * 16 + l15;
        const float bv = bias[n];
#pragma unroll
        for (int r = 0; r < 8; ++r) {
            out[(size_t)(mBase + r) * N_TOT + n] = accs[t][r] * s + bv;
        }
    }
}

extern "C" void kernel_launch(void* const* d_in, const int* in_sizes, int n_in,
                              void* d_out, int out_size, void* d_ws, size_t ws_size,
                              hipStream_t stream) {
    (void)in_sizes; (void)n_in; (void)out_size; (void)ws_size;
    const float* inp  = (const float*)d_in[0];
    const float* wgt  = (const float*)d_in[1];
    const float* bias = (const float*)d_in[2];
    float* out = (float*)d_out;

    char* ws = (char*)d_ws;
    float* scales  = (float*)ws;
    float* partIn  = (float*)(ws + 256);
    float* partW   = (float*)(ws + 256 + 4096);
    unsigned char* A8 = (unsigned char*)(ws + 16384);
    unsigned char* W8 = A8 + (size_t)M_TOT * K_TOT;

    const int nIn = M_TOT * K_TOT;
    const int nW  = N_TOT * K_TOT;

    lb_absum_partial<<<1024, 256, 0, stream>>>(inp, partIn, nIn / 4);
    lb_finalize_scale<<<1, 256, 0, stream>>>(partIn, scales + 0, 1.0f / (float)nIn);
    lb_absum_partial<<<1024, 256, 0, stream>>>(wgt, partW, nW / 4);
    lb_finalize_scale<<<1, 256, 0, stream>>>(partW, scales + 1, 1.0f / (float)nW);

    lb_quant_fp8<<<nIn / 4 / 256, 256, 0, stream>>>(inp, A8, scales + 0, nIn / 4);
    lb_quant_fp8<<<nW  / 4 / 256, 256, 0, stream>>>(wgt, W8, scales + 1, nW / 4);

    lb_gemm_fp8<<<(M_TOT / 16) * (N_TOT / 64) / 8, 256, 0, stream>>>(
        A8, W8, bias, scales, out);
}
